// MultiHeadSelfAttention_26697516712001
// MI455X (gfx1250) — matrix operand order repaired and checked by reading, compiled, not run on hardware
//
#include <hip/hip_runtime.h>

// ---------------------------------------------------------------------------
// Types
// ---------------------------------------------------------------------------
typedef __bf16 bf16;
typedef bf16  v16bf __attribute__((ext_vector_type(16)));
typedef bf16  v8bf  __attribute__((ext_vector_type(8)));
typedef float v8f   __attribute__((ext_vector_type(8)));

#define B_DIM 2
#define N_SEQ 2048
#define C_DIM 1024
#define H_NUM 16
#define D_HEAD 64
#define M_ROWS (B_DIM * N_SEQ)   // 4096

static __device__ inline v8f vzero() {
    v8f z = {0.f, 0.f, 0.f, 0.f, 0.f, 0.f, 0.f, 0.f};
    return z;
}

static __device__ inline v16bf ld_frag2(const bf16* p0, const bf16* p1) {
    union { v16bf v; v8bf h[2]; } u;
    u.h[0] = *(const v8bf*)p0;
    u.h[1] = *(const v8bf*)p1;
    return u.v;
}

static __device__ inline v8f wmma_bf16(v16bf a, v16bf b, v8f c) {
    // D = A(16x32 bf16) * B(32x16 bf16) + C(16x16 f32)
    return __builtin_amdgcn_wmma_f32_16x16x32_bf16(
        false, a, false, b, (short)0, c, false, false);
}

// ---------------------------------------------------------------------------
// Kernel 1: f32 -> bf16 convert
// ---------------------------------------------------------------------------
__global__ __launch_bounds__(256) void cvt_f32_to_bf16(
    const float* __restrict__ in, bf16* __restrict__ out, int n) {
    int i = blockIdx.x * blockDim.x + threadIdx.x;
    if (i < n) out[i] = (bf16)in[i];
}

// ---------------------------------------------------------------------------
// Kernel 2: GEMM  C[m,o] = sum_k A[m,k] * W[o,k]   (bf16 in, bf16 out)
// One wave computes a 32x64 tile (2 A-frags reuse each B-frag). Block = 8 waves.
// ---------------------------------------------------------------------------
__global__ __launch_bounds__(256) void gemm_bf16(
    const bf16* __restrict__ A, const bf16* __restrict__ W,
    bf16* __restrict__ C, int M, int Nc, int K) {
    const int lane = threadIdx.x & 31;
    const int wave = threadIdx.x >> 5;
    const int half = lane >> 4;
    const int l16  = lane & 15;

    const int ntilesN = Nc >> 6;
    const int wid = blockIdx.x * 8 + wave;
    const int mt = wid / ntilesN;              // 32-row tile index
    const int nt = wid % ntilesN;              // 64-col tile index
    if (mt * 32 >= M) return;

    const bf16* arow0 = A + (size_t)(mt * 32 + l16) * K;
    const bf16* arow1 = arow0 + (size_t)16 * K;

    v8f acc0[4], acc1[4];
    #pragma unroll
    for (int j = 0; j < 4; ++j) { acc0[j] = vzero(); acc1[j] = vzero(); }

    for (int k0 = 0; k0 < K; k0 += 32) {
        v16bf af0 = ld_frag2(arow0 + k0 + half * 16, arow0 + k0 + half * 16 + 8);
        v16bf af1 = ld_frag2(arow1 + k0 + half * 16, arow1 + k0 + half * 16 + 8);
        #pragma unroll
        for (int j = 0; j < 4; ++j) {
            const bf16* wrow = W + (size_t)(nt * 64 + j * 16 + l16) * K + k0 + half * 16;
            v16bf bfv = *(const v16bf*)wrow;   // 16 contiguous bf16 (32B)
            acc0[j] = wmma_bf16(af0, bfv, acc0[j]);
            acc1[j] = wmma_bf16(af1, bfv, acc1[j]);
        }
    }

    #pragma unroll
    for (int j = 0; j < 4; ++j) {
        #pragma unroll
        for (int r = 0; r < 8; ++r) {
            const int n  = nt * 64 + j * 16 + l16;
            const int m0 = mt * 32 + half * 8 + r;
            const int m1 = m0 + 16;
            C[(size_t)m0 * Nc + n] = (bf16)acc0[j][r];
            C[(size_t)m1 * Nc + n] = (bf16)acc1[j][r];
        }
    }
}

// ---------------------------------------------------------------------------
// Kernel 3: repack V transposed:  Vt[b,h,d,n] = Qkv[b*N+n, 2C + h*D + d]
// ---------------------------------------------------------------------------
__global__ __launch_bounds__(256) void repack_vt(
    const bf16* __restrict__ Qkv, bf16* __restrict__ Vt) {
    const int idx = blockIdx.x * blockDim.x + threadIdx.x; // over B*H*N*D, d inner
    const int d  = idx & 63;
    const int t  = idx >> 6;
    const int n  = t & (N_SEQ - 1);
    const int bh = t >> 11;
    const int b  = bh >> 4;
    const int h  = bh & 15;
    const bf16 v = Qkv[(size_t)(b * N_SEQ + n) * (3 * C_DIM) + 2 * C_DIM + h * D_HEAD + d];
    Vt[((size_t)bh * D_HEAD + d) * N_SEQ + n] = v;
}

// ---------------------------------------------------------------------------
// Kernel 4: flash attention, transposed-score formulation.
//   S^T = K * Q^T   (C-frag: lane = fixed query column -> in-lane softmax)
//   O^T = V^T * P^T (A = Vt rows, B = P^T staged via one LDS round trip)
// One wave: 16 query rows for one (b,h), loops over key tiles of 32.
// ---------------------------------------------------------------------------
__global__ __launch_bounds__(256) void attn_flash(
    const bf16* __restrict__ Qkv, const bf16* __restrict__ Vt,
    bf16* __restrict__ O) {
    __shared__ bf16 Psh[8][16][40];            // per-wave P^T tile [query][key], padded

    const int lane = threadIdx.x & 31;
    const int wave = threadIdx.x >> 5;
    const int half = lane >> 4;
    const int l16  = lane & 15;

    const int wid = blockIdx.x * 8 + wave;     // 0..4095
    const int qt  = wid & 127;                 // query tile: N/16
    const int bh  = wid >> 7;                  // 0..31
    const int b   = bh >> 4;
    const int h   = bh & 15;

    const size_t rs = 3 * C_DIM;               // Qkv row stride
    const bf16* Qbase = Qkv + (size_t)(b * N_SEQ) * rs + h * D_HEAD;
    const bf16* Kbase = Qbase + C_DIM;
    const bf16* Vbase = Vt + (size_t)bh * D_HEAD * N_SEQ;

    // Q as B-fragments (Q^T chunks, 32d x 16q): lane = query col l16, K = d
    const bf16* qrow = Qbase + (size_t)(qt * 16 + l16) * rs;
    const v16bf qb0 = ld_frag2(qrow + half * 8, qrow + 16 + half * 8);        // d 0..31
    const v16bf qb1 = ld_frag2(qrow + 32 + half * 8, qrow + 48 + half * 8);   // d 32..63

    v8f acc[4];                                // O^T: rows d = j*16+8*half+r, col q=l16
    #pragma unroll
    for (int j = 0; j < 4; ++j) acc[j] = vzero();
    float rmax = -1e30f, rsum = 0.f;           // per-query (per-lane) scalars

    const float scale = 0.125f;                // 1/sqrt(64)

    for (int kt = 0; kt < N_SEQ; kt += 32) {
        // prefetch next tile's K rows / V row (wrapped to stay in range)
        {
            const int ktn = (kt + 32) & (N_SEQ - 1);
            __builtin_prefetch(Kbase + (size_t)(ktn + l16) * rs, 0, 0);
            __builtin_prefetch(Kbase + (size_t)(ktn + 16 + l16) * rs, 0, 0);
            __builtin_prefetch(Vbase + (size_t)l16 * N_SEQ + ktn, 0, 0);
        }

        // ---- S^T[32 keys x 16 queries] = K * Q^T ----
        const bf16* kr0 = Kbase + (size_t)(kt + l16) * rs;        // keys kt..kt+15
        const bf16* kr1 = Kbase + (size_t)(kt + 16 + l16) * rs;   // keys kt+16..kt+31
        v16bf ka0 = ld_frag2(kr0 + half * 8,      kr0 + 16 + half * 8);  // d 0..31
        v16bf ka1 = ld_frag2(kr0 + 32 + half * 8, kr0 + 48 + half * 8);  // d 32..63
        v16bf kb0 = ld_frag2(kr1 + half * 8,      kr1 + 16 + half * 8);
        v16bf kb1 = ld_frag2(kr1 + 32 + half * 8, kr1 + 48 + half * 8);
        v8f stA = vzero(), stB = vzero();
        stA = wmma_bf16(ka0, qb0, stA);
        stA = wmma_bf16(ka1, qb1, stA);        // keys kt + 8*half + r
        stB = wmma_bf16(kb0, qb0, stB);
        stB = wmma_bf16(kb1, qb1, stB);        // keys kt+16 + 8*half + r

        // ---- online softmax: all 16 key-scores for query l16 are in-lane ----
        float tm = fmaxf(stA[0], stB[0]);
        #pragma unroll
        for (int r = 1; r < 8; ++r) tm = fmaxf(tm, fmaxf(stA[r], stB[r]));
        tm = fmaxf(tm, __shfl_xor(tm, 16, 32));          // merge the two halves
        const float nm   = fmaxf(rmax, tm);
        const float corr = __expf((rmax - nm) * scale);
        rmax = nm;

        float ts = 0.f;
        v8bf pa, pb;
        #pragma unroll
        for (int r = 0; r < 8; ++r) {
            const float p0 = __expf((stA[r] - nm) * scale);
            const float p1 = __expf((stB[r] - nm) * scale);
            ts += p0 + p1;
            pa[r] = (bf16)p0;
            pb[r] = (bf16)p1;
        }
        ts += __shfl_xor(ts, 16, 32);
        rsum = rsum * corr + ts;

        #pragma unroll
        for (int j = 0; j < 4; ++j)
            #pragma unroll
            for (int r = 0; r < 8; ++r) acc[j][r] *= corr;

        // ---- stage P^T in LDS: Psh[query][key], both sides contiguous ----
        *(v8bf*)&Psh[wave][l16][8 * half]      = pa;   // keys 8*half+0..7
        *(v8bf*)&Psh[wave][l16][16 + 8 * half] = pb;   // keys 16+8*half+0..7
        asm volatile("s_wait_dscnt 0" ::: "memory");   // intra-wave DS RAW
        // B-frag of P^T: lane = query col l16, K = keys half*16+0..15
        v16bf pf = *(const v16bf*)&Psh[wave][l16][half * 16];

        // ---- O^T += V^T * P^T : A = Vt rows (d = j*16 + l16) ----
        #pragma unroll
        for (int j = 0; j < 4; ++j) {
            const bf16* vrow = Vbase + (size_t)(j * 16 + l16) * N_SEQ + kt;
            v16bf vf = ld_frag2(vrow + half * 16, vrow + half * 16 + 8);
            acc[j] = wmma_bf16(vf, pf, acc[j]);
        }
    }

    // ---- finalize: O[b, n, h*D + d], 8 contiguous d per lane -> 16B stores ----
    const float rinv = 1.0f / rsum;
    const int n = qt * 16 + l16;
    #pragma unroll
    for (int j = 0; j < 4; ++j) {
        v8bf ov;
        #pragma unroll
        for (int r = 0; r < 8; ++r) ov[r] = (bf16)(acc[j][r] * rinv);
        *(v8bf*)&O[(size_t)(b * N_SEQ + n) * C_DIM + h * D_HEAD + j * 16 + 8 * half] = ov;
    }
}

// ---------------------------------------------------------------------------
// Kernel 5: output projection  Y[m,o] = sum_c O[m,c]*Wout[o,c] + bias[o] (f32 out)
// 32x64 wave tile, bias folded into accumulator init.
// ---------------------------------------------------------------------------
__global__ __launch_bounds__(256) void gemm_bias_f32(
    const bf16* __restrict__ A, const bf16* __restrict__ W,
    const float* __restrict__ bias, float* __restrict__ C,
    int M, int Nc, int K) {
    const int lane = threadIdx.x & 31;
    const int wave = threadIdx.x >> 5;
    const int half = lane >> 4;
    const int l16  = lane & 15;

    const int ntilesN = Nc >> 6;
    const int wid = blockIdx.x * 8 + wave;
    const int mt = wid / ntilesN;
    const int nt = wid % ntilesN;
    if (mt * 32 >= M) return;

    const bf16* arow0 = A + (size_t)(mt * 32 + l16) * K;
    const bf16* arow1 = arow0 + (size_t)16 * K;

    v8f acc0[4], acc1[4];
    #pragma unroll
    for (int j = 0; j < 4; ++j) {
        const float bv = bias[nt * 64 + j * 16 + l16];
        #pragma unroll
        for (int r = 0; r < 8; ++r) { acc0[j][r] = bv; acc1[j][r] = bv; }
    }

    for (int k0 = 0; k0 < K; k0 += 32) {
        v16bf af0 = ld_frag2(arow0 + k0 + half * 16, arow0 + k0 + half * 16 + 8);
        v16bf af1 = ld_frag2(arow1 + k0 + half * 16, arow1 + k0 + half * 16 + 8);
        #pragma unroll
        for (int j = 0; j < 4; ++j) {
            const bf16* wrow = W + (size_t)(nt * 64 + j * 16 + l16) * K + k0 + half * 16;
            v16bf bfv = *(const v16bf*)wrow;
            acc0[j] = wmma_bf16(af0, bfv, acc0[j]);
            acc1[j] = wmma_bf16(af1, bfv, acc1[j]);
        }
    }

    #pragma unroll
    for (int j = 0; j < 4; ++j) {
        #pragma unroll
        for (int r = 0; r < 8; ++r) {
            const int n  = nt * 64 + j * 16 + l16;
            const int m0 = mt * 32 + half * 8 + r;
            const int m1 = m0 + 16;
            C[(size_t)m0 * Nc + n] = acc0[j][r];
            C[(size_t)m1 * Nc + n] = acc1[j][r];
        }
    }
}

// ---------------------------------------------------------------------------
// Launch
// ---------------------------------------------------------------------------
extern "C" void kernel_launch(void* const* d_in, const int* in_sizes, int n_in,
                              void* d_out, int out_size, void* d_ws, size_t ws_size,
                              hipStream_t stream) {
    const float* x     = (const float*)d_in[0];   // [B,N,C]
    const float* w_qkv = (const float*)d_in[1];   // [3C,C]
    const float* w_out = (const float*)d_in[2];   // [C,C]
    const float* b_out = (const float*)d_in[3];   // [C]
    float* y = (float*)d_out;                     // [B,N,C]

    char* ws = (char*)d_ws;
    size_t off = 0;
    auto carve = [&](size_t bytes) {
        char* p = ws + off;
        off += (bytes + 255) & ~size_t(255);
        return p;
    };
    bf16* xb    = (bf16*)carve((size_t)M_ROWS * C_DIM * 2);                // 8 MB
    bf16* wqkvb = (bf16*)carve((size_t)3 * C_DIM * C_DIM * 2);             // 6 MB
    bf16* woutb = (bf16*)carve((size_t)C_DIM * C_DIM * 2);                 // 2 MB
    bf16* qkv   = (bf16*)carve((size_t)M_ROWS * 3 * C_DIM * 2);            // 24 MB
    bf16* vt    = (bf16*)carve((size_t)B_DIM * H_NUM * D_HEAD * N_SEQ * 2);// 8 MB
    bf16* obuf  = (bf16*)carve((size_t)M_ROWS * C_DIM * 2);                // 8 MB
    (void)ws_size; (void)in_sizes; (void)n_in; (void)out_size;

    const int nx  = M_ROWS * C_DIM;       // 4194304
    const int nwq = 3 * C_DIM * C_DIM;    // 3145728
    const int nwo = C_DIM * C_DIM;        // 1048576

    cvt_f32_to_bf16<<<nx  / 256, 256, 0, stream>>>(x, xb, nx);
    cvt_f32_to_bf16<<<nwq / 256, 256, 0, stream>>>(w_qkv, wqkvb, nwq);
    cvt_f32_to_bf16<<<nwo / 256, 256, 0, stream>>>(w_out, woutb, nwo);

    // QKV GEMM: (4096/32)*(3072/64) = 6144 wave tiles / 8 waves
    gemm_bf16<<<(128 * 48) / 8, 256, 0, stream>>>(xb, wqkvb, qkv,
                                                  M_ROWS, 3 * C_DIM, C_DIM);

    // V transpose repack
    const int nvt = B_DIM * H_NUM * N_SEQ * D_HEAD;
    repack_vt<<<nvt / 256, 256, 0, stream>>>(qkv, vt);

    // flash attention: B*H*(N/16) = 4096 waves -> 512 blocks
    attn_flash<<<(B_DIM * H_NUM * (N_SEQ / 16)) / 8, 256, 0, stream>>>(qkv, vt, obuf);

    // output projection: (4096/32)*(1024/64) = 2048 wave tiles / 8 waves
    gemm_bias_f32<<<(128 * 16) / 8, 256, 0, stream>>>(obuf, woutb, b_out, y,
                                                      M_ROWS, C_DIM, C_DIM);
}
